// mLSTMLayer_78537771975161
// MI455X (gfx1250) — matrix operand order repaired and checked by reading, not compiled, not run on hardware
//
#include <hip/hip_runtime.h>
#include <math.h>

#define B_ 2
#define S_ 2048
#define FEAT_ 1024
#define HID_ 2048
#define NH_ 4
#define DH_ 512
#define KK_ 4
#define QSCALE 0.04419417382415922f   /* 1/sqrt(512) */

typedef __attribute__((ext_vector_type(16))) __bf16 v16bf;
typedef __attribute__((ext_vector_type(8)))  float  v8f;

union Frag16 { v16bf v; unsigned u[8]; };
union FragF  { v8f  v; float    f[8]; };

static __device__ __forceinline__ unsigned short f2bf(float f) {
  union { float f; unsigned u; } x; x.f = f;
  unsigned r = (x.u + 0x7FFFu + ((x.u >> 16) & 1u)) >> 16;
  return (unsigned short)r;
}
static __device__ __forceinline__ unsigned pack_bf2(float lo, float hi) {
  return (unsigned)f2bf(lo) | ((unsigned)f2bf(hi) << 16);
}
static __device__ __forceinline__ float bf2f(unsigned short b) {
  union { unsigned u; float f; } x; x.u = ((unsigned)b) << 16;
  return x.f;
}
// A-fragment dword (2xbf16) index within a 16-pair row, per ISA 16-bit A layout
static __device__ __forceinline__ int pairA(int i, int hf) {
  return (i < 4 ? i : i + 4) + hf * 4;
}
static __device__ __forceinline__ float siluf(float x) {
  return x / (1.f + expf(-x));
}

// ---- CDNA5 async global->LDS copy (ASYNCcnt tracked), 16 bytes per lane ----
static __device__ __forceinline__ void async_copy16(void* lds, const void* g) {
  unsigned lo = (unsigned)(unsigned long long)lds;       // LDS offset = addr[31:0]
  unsigned long long ga = (unsigned long long)g;
  asm volatile("global_load_async_to_lds_b128 %0, %1, off"
               :: "v"(lo), "v"(ga) : "memory");
}
static __device__ __forceinline__ void async_wait0() {
  asm volatile("s_wait_asynccnt 0x0" ::: "memory");
}

// ---------------------------------------------------------------------------
// Weight pre-pass: WT[n][k] = bf16(W[k][n]);  W is K x N row-major.
// ---------------------------------------------------------------------------
__global__ __launch_bounds__(256) void transpose_bf16_kernel(
    const float* __restrict__ W, unsigned short* __restrict__ WT, int K, int N) {
  size_t idx = (size_t)blockIdx.x * 256 + threadIdx.x;   // over N*K
  int n = (int)(idx / K), k = (int)(idx % K);
  WT[idx] = f2bf(W[(size_t)k * N + n]);
}

// ---------------------------------------------------------------------------
// GEMM: C[M,N] = A[M,K](f32) @ B, with B given pre-transposed bf16 BT[N,K].
// A staged with f32->bf16 convert; B staged with async global->LDS copies.
// ---------------------------------------------------------------------------
__global__ __launch_bounds__(256) void wmma_gemm_f32a(
    const float* __restrict__ A, const unsigned short* __restrict__ BT,
    float* __restrict__ C, int M, int N, int K) {
  __shared__ unsigned As[128][16];   // [m][k-pair]
  __shared__ unsigned Bs[128][16];   // [n][k-pair]
  const int tid = threadIdx.x;
  const int lane = tid & 31, wid = tid >> 5;
  const int hf = lane >> 4, ln = lane & 15;
  const int wrow = wid >> 2, wcol = wid & 3;
  const int rowBase = blockIdx.y * 128;
  const int colBase = blockIdx.x * 128;

  FragF acc[4][2];
#pragma unroll
  for (int mf = 0; mf < 4; ++mf)
#pragma unroll
    for (int nf = 0; nf < 2; ++nf)
#pragma unroll
      for (int r = 0; r < 8; ++r) acc[mf][nf].f[r] = 0.f;

  const int nk = K / 32;
  for (int kt = 0; kt < nk; ++kt) {
    __syncthreads();
    if (kt + 1 < nk) {
      __builtin_prefetch(A + (size_t)(rowBase + (tid >> 1)) * K + (kt + 1) * 32, 0, 1);
      __builtin_prefetch(BT + (size_t)(colBase + (tid >> 1)) * K + (kt + 1) * 32, 0, 1);
    }
    // B tile: 128 rows x 64B, async copy (2 chunks of 16B per thread)
#pragma unroll
    for (int c = tid; c < 512; c += 256) {
      int n = c >> 2, seg = c & 3;
      async_copy16((char*)Bs + n * 64 + seg * 16,
                   BT + (size_t)(colBase + n) * K + kt * 32 + seg * 8);
    }
    // A tile: convert f32 -> bf16 pairs
#pragma unroll
    for (int p = tid; p < 2048; p += 256) {
      int r = p >> 4, kp = p & 15;
      const float* ap = A + (size_t)(rowBase + r) * K + kt * 32 + kp * 2;
      As[r][kp] = pack_bf2(ap[0], ap[1]);
    }
    async_wait0();
    __syncthreads();

    Frag16 bfr[2];
#pragma unroll
    for (int nf = 0; nf < 2; ++nf) {
      int n = wcol * 32 + nf * 16 + ln;
#pragma unroll
      for (int i = 0; i < 8; ++i) bfr[nf].u[i] = Bs[n][i + hf * 8];
    }
#pragma unroll
    for (int mf = 0; mf < 4; ++mf) {
      int m = wrow * 64 + mf * 16 + ln;
      Frag16 afr;
#pragma unroll
      for (int i = 0; i < 8; ++i) afr.u[i] = As[m][i + hf * 8];
#pragma unroll
      for (int nf = 0; nf < 2; ++nf)
        acc[mf][nf].v = __builtin_amdgcn_wmma_f32_16x16x32_bf16(
            false, afr.v, false, bfr[nf].v, (short)0, acc[mf][nf].v, false, false);
    }
  }

#pragma unroll
  for (int mf = 0; mf < 4; ++mf)
#pragma unroll
    for (int nf = 0; nf < 2; ++nf)
#pragma unroll
      for (int r = 0; r < 8; ++r) {
        int m = rowBase + wrow * 64 + mf * 16 + r + 8 * hf;
        int n = colBase + wcol * 32 + nf * 16 + ln;
        C[(size_t)m * N + n] = acc[mf][nf].f[r];
      }
}

// ---------------------------------------------------------------------------
// GEMM: C[M,N] = A[M,K](bf16) @ B, BT[N,K] bf16.  Both tiles async-staged.
// ---------------------------------------------------------------------------
__global__ __launch_bounds__(256) void wmma_gemm_bf16(
    const unsigned short* __restrict__ A, const unsigned short* __restrict__ BT,
    float* __restrict__ C, int M, int N, int K) {
  __shared__ unsigned As[128][16];
  __shared__ unsigned Bs[128][16];
  const int tid = threadIdx.x;
  const int lane = tid & 31, wid = tid >> 5;
  const int hf = lane >> 4, ln = lane & 15;
  const int wrow = wid >> 2, wcol = wid & 3;
  const int rowBase = blockIdx.y * 128;
  const int colBase = blockIdx.x * 128;

  FragF acc[4][2];
#pragma unroll
  for (int mf = 0; mf < 4; ++mf)
#pragma unroll
    for (int nf = 0; nf < 2; ++nf)
#pragma unroll
      for (int r = 0; r < 8; ++r) acc[mf][nf].f[r] = 0.f;

  const int nk = K / 32;
  for (int kt = 0; kt < nk; ++kt) {
    __syncthreads();
    if (kt + 1 < nk) {
      __builtin_prefetch(A + (size_t)(rowBase + (tid >> 1)) * K + (kt + 1) * 32, 0, 1);
      __builtin_prefetch(BT + (size_t)(colBase + (tid >> 1)) * K + (kt + 1) * 32, 0, 1);
    }
#pragma unroll
    for (int c = tid; c < 512; c += 256) {
      int m = c >> 2, seg = c & 3;
      async_copy16((char*)As + m * 64 + seg * 16,
                   A + (size_t)(rowBase + m) * K + kt * 32 + seg * 8);
    }
#pragma unroll
    for (int c = tid; c < 512; c += 256) {
      int n = c >> 2, seg = c & 3;
      async_copy16((char*)Bs + n * 64 + seg * 16,
                   BT + (size_t)(colBase + n) * K + kt * 32 + seg * 8);
    }
    async_wait0();
    __syncthreads();

    Frag16 bfr[2];
#pragma unroll
    for (int nf = 0; nf < 2; ++nf) {
      int n = wcol * 32 + nf * 16 + ln;
#pragma unroll
      for (int i = 0; i < 8; ++i) bfr[nf].u[i] = Bs[n][i + hf * 8];
    }
#pragma unroll
    for (int mf = 0; mf < 4; ++mf) {
      int m = wrow * 64 + mf * 16 + ln;
      Frag16 afr;
#pragma unroll
      for (int i = 0; i < 8; ++i) afr.u[i] = As[m][i + hf * 8];
#pragma unroll
      for (int nf = 0; nf < 2; ++nf)
        acc[mf][nf].v = __builtin_amdgcn_wmma_f32_16x16x32_bf16(
            false, afr.v, false, bfr[nf].v, (short)0, acc[mf][nf].v, false, false);
    }
  }

#pragma unroll
  for (int mf = 0; mf < 4; ++mf)
#pragma unroll
    for (int nf = 0; nf < 2; ++nf)
#pragma unroll
      for (int r = 0; r < 8; ++r) {
        int m = rowBase + wrow * 64 + mf * 16 + r + 8 * hf;
        int n = colBase + wcol * 32 + nf * 16 + ln;
        C[(size_t)m * N + n] = acc[mf][nf].f[r];
      }
}

// ---------------------------------------------------------------------------
// Fused causal conv(K=4)+SiLU, block-diagonal q/k/v, gate pre-activations.
// ---------------------------------------------------------------------------
__global__ __launch_bounds__(256) void conv_qkv_gates(
    const float* __restrict__ up, const float* __restrict__ conv_state,
    const float* __restrict__ conv_w, const float* __restrict__ conv_b,
    const float* __restrict__ Wq, const float* __restrict__ Wk,
    const float* __restrict__ Wv,
    const float* __restrict__ wi_k, const float* __restrict__ wi_b,
    const float* __restrict__ wf_k, const float* __restrict__ wf_b,
    float* __restrict__ conv_act,
    unsigned short* __restrict__ qbf, unsigned short* __restrict__ kbf,
    unsigned short* __restrict__ vbf,
    float* __restrict__ ig_pre, float* __restrict__ fg_pre) {
  __shared__ float red[256 * 8];
  const int t  = blockIdx.x;
  const int b  = t / S_, tt = t % S_;
  const int tid = threadIdx.x;
  const int c0 = tid * 8;

  float xw[4][8];
#pragma unroll
  for (int j = 0; j < 4; ++j) {
    int idx = tt + 1 + j;            // index into xc = [conv_state(4) | x]
#pragma unroll
    for (int u = 0; u < 8; ++u) {
      int c = c0 + u;
      xw[j][u] = (idx < KK_)
          ? conv_state[((size_t)b * KK_ + idx) * HID_ + c]
          : up[((size_t)b * S_ + (idx - KK_)) * (2 * HID_) + c];
    }
  }
  float ca[8], xv[8];
#pragma unroll
  for (int u = 0; u < 8; ++u) {
    int c = c0 + u;
    float cv = conv_b[c];
#pragma unroll
    for (int j = 0; j < 4; ++j) cv += conv_w[j * HID_ + c] * xw[j][u];
    ca[u] = siluf(cv);
    xv[u] = xw[3][u];
  }

  float qv[8], kv[8], vv[8];
#pragma unroll
  for (int blk = 0; blk < 2; ++blk) {
    int n = (c0 >> 2) + blk;
    const float* wq = Wq + n * 16;
    const float* wk = Wk + n * 16;
    const float* wv = Wv + n * 16;
#pragma unroll
    for (int o = 0; o < 4; ++o) {
      float aq = 0.f, ak = 0.f, av = 0.f;
#pragma unroll
      for (int i = 0; i < 4; ++i) {
        aq += ca[blk * 4 + i] * wq[i * 4 + o];
        ak += ca[blk * 4 + i] * wk[i * 4 + o];
        av += xv[blk * 4 + i] * wv[i * 4 + o];
      }
      qv[blk * 4 + o] = aq; kv[blk * 4 + o] = ak; vv[blk * 4 + o] = av;
    }
  }

  const size_t base = (size_t)t * HID_ + c0;
#pragma unroll
  for (int u = 0; u < 8; ++u) {
    conv_act[base + u] = ca[u];
    qbf[base + u] = f2bf(qv[u] * QSCALE);
    kbf[base + u] = f2bf(kv[u]);
    vbf[base + u] = f2bf(vv[u]);
  }

  float pi[4] = {0, 0, 0, 0}, pf[4] = {0, 0, 0, 0};
#pragma unroll
  for (int u = 0; u < 8; ++u) {
    int c = c0 + u;
#pragma unroll
    for (int h = 0; h < 4; ++h) {
      pi[h] += qv[u] * wi_k[(size_t)c * 4 + h]
             + kv[u] * wi_k[(size_t)(HID_ + c) * 4 + h]
             + vv[u] * wi_k[(size_t)(2 * HID_ + c) * 4 + h];
      pf[h] += qv[u] * wf_k[(size_t)c * 4 + h]
             + kv[u] * wf_k[(size_t)(HID_ + c) * 4 + h]
             + vv[u] * wf_k[(size_t)(2 * HID_ + c) * 4 + h];
    }
  }
#pragma unroll
  for (int h = 0; h < 4; ++h) { red[tid * 8 + h] = pi[h]; red[tid * 8 + 4 + h] = pf[h]; }
  for (int s2 = 128; s2 > 0; s2 >>= 1) {
    __syncthreads();
    if (tid < s2)
#pragma unroll
      for (int u = 0; u < 8; ++u) red[tid * 8 + u] += red[(tid + s2) * 8 + u];
  }
  __syncthreads();
  if (tid < 4) {
    ig_pre[(size_t)t * NH_ + tid] = red[tid]     + wi_b[tid];
    fg_pre[(size_t)t * NH_ + tid] = red[4 + tid] + wf_b[tid];
  }
}

// ---------------------------------------------------------------------------
// Per-(b,h) scans: lfc = cumsum(-softplus(-f_gate)); ep = cumsum(mask)
// ---------------------------------------------------------------------------
__global__ __launch_bounds__(32) void scan_kernel(
    const float* __restrict__ fg_pre, const int* __restrict__ mask,
    float* __restrict__ lfc, int* __restrict__ ep) {
  const int id = blockIdx.x, b = id / NH_, h = id % NH_;
  const int lane = threadIdx.x;
  const float* src = fg_pre + (size_t)b * S_ * NH_ + (size_t)h * S_;
  float* dst = lfc + ((size_t)b * NH_ + h) * S_;
  float carry = 0.f;
  for (int c = 0; c < S_ / 32; ++c) {
    float x = src[c * 32 + lane];
    float lf = (x >= 0.f) ? -log1pf(expf(-x)) : (x - log1pf(expf(x)));
    for (int off = 1; off < 32; off <<= 1) {
      float n = __shfl_up(lf, off, 32);
      if (lane >= off) lf += n;
    }
    dst[c * 32 + lane] = carry + lf;
    carry += __shfl(lf, 31, 32);
  }
  if (h == 0) {
    const int* ms = mask + (size_t)b * S_;
    int* ed = ep + (size_t)b * S_;
    int ic = 0;
    for (int c = 0; c < S_ / 32; ++c) {
      int v = ms[c * 32 + lane];
      for (int off = 1; off < 32; off <<= 1) {
        int n = __shfl_up(v, off, 32);
        if (lane >= off) v += n;
      }
      ed[c * 32 + lane] = ic + v;
      ic += __shfl(v, 31, 32);
    }
  }
}

// ---------------------------------------------------------------------------
// qn[b,h,t] = qh[b,h,t,:] . n0[b,h,:]   (one wave per row)
// ---------------------------------------------------------------------------
__global__ __launch_bounds__(256) void qn_kernel(
    const unsigned short* __restrict__ qbf, const float* __restrict__ n0,
    float* __restrict__ qn) {
  const int row = blockIdx.x * 8 + (threadIdx.x >> 5);
  const int lane = threadIdx.x & 31;
  const int bh = row / S_;
  const unsigned short* q = qbf + (size_t)row * DH_;
  const float* n0p = n0 + (size_t)bh * DH_;
  float s = 0.f;
  for (int i = 0; i < DH_ / 32; ++i) {
    int d = i * 32 + lane;
    s += bf2f(q[d]) * n0p[d];
  }
  for (int off = 16; off; off >>= 1) s += __shfl_xor(s, off, 32);
  if (lane == 0) qn[row] = s;
}

// ---------------------------------------------------------------------------
// Flash-style exact mLSTM attention.  grid = (S/64, B*NH, 2 dh-halves),
// block = 128 (4 waves x 16 rows).  WMMA for qC0, QK^T and PV.
// Q/K tiles staged via async global->LDS (bf16 source), V transposed manually.
// ---------------------------------------------------------------------------
__global__ __launch_bounds__(128) void attn_kernel(
    const unsigned short* __restrict__ qbf, const unsigned short* __restrict__ kbf,
    const unsigned short* __restrict__ vbf,
    const float* __restrict__ lfcg, const float* __restrict__ gIg,
    const int* __restrict__ epg, const float* __restrict__ qng,
    const float* __restrict__ C0g, const float* __restrict__ m0g,
    float* __restrict__ ht) {
  __shared__ unsigned smQ[64 * 64];        // Q chunk [i][d-pair], 16KB
  __shared__ unsigned smKV[8192];          // Ks / VTs / C0T overlay, 32KB
  __shared__ unsigned short smP[4 * 16 * 64];
  __shared__ float lfc_row[64], gi_row[64], qn_row[64], lfc_col[64];
  __shared__ int ep_row[64], ep_col[64];

  const int tid = threadIdx.x, lane = tid & 31, w = tid >> 5;
  const int hf = lane >> 4, ln = lane & 15;
  const int row0 = blockIdx.x * 64;
  const int bh = blockIdx.y, b = bh / NH_, h = bh % NH_;
  const int dhalf = blockIdx.z * 256;

  const size_t baseRow = (size_t)bh * S_;
  const unsigned short* qb = qbf + baseRow * DH_;
  const unsigned short* kb = kbf + baseRow * DH_;
  const unsigned short* vb = vbf + baseRow * DH_;
  const float* C0p = C0g + (size_t)bh * DH_ * DH_;
  const float m0v = m0g[bh];

  for (int i = tid; i < 64; i += 128) {
    lfc_row[i] = lfcg[baseRow + row0 + i];
    gi_row[i]  = gIg[(size_t)b * S_ * NH_ + (size_t)h * S_ + row0 + i];
    qn_row[i]  = qng[baseRow + row0 + i];
    ep_row[i]  = epg[(size_t)b * S_ + row0 + i];
  }

  FragF acc[16];
#pragma unroll
  for (int nf = 0; nf < 16; ++nf)
#pragma unroll
    for (int r = 0; r < 8; ++r) acc[nf].f[r] = 0.f;

  // ---- init acc with qC0 (implicit running max = sdc) ----
  for (int kc = 0; kc < DH_; kc += 64) {
    __syncthreads();
    // Q chunk rows (64 x 128B), async
    for (int c = tid; c < 512; c += 128) {
      int i = c >> 3, seg = c & 7;
      async_copy16((char*)smQ + i * 256 + seg * 16,
                   qb + (size_t)(row0 + i) * DH_ + kc + seg * 8);
    }
    // C0T [n 256][k 64] bf16, convert + transpose
    for (int p = tid; p < 256 * 32; p += 128) {
      int n = p >> 5, kp = p & 31;
      const float* cp = C0p + (size_t)(kc + kp * 2) * DH_ + dhalf + n;
      smKV[n * 32 + kp] = pack_bf2(cp[0], cp[DH_]);
    }
    async_wait0();
    __syncthreads();
#pragma unroll
    for (int ks = 0; ks < 2; ++ks) {
      Frag16 afr;
      int m = w * 16 + ln;
#pragma unroll
      for (int i = 0; i < 8; ++i) afr.u[i] = smQ[m * 64 + ks * 16 + i + hf * 8];
#pragma unroll
      for (int nf = 0; nf < 16; ++nf) {
        Frag16 bfr;
        int n = nf * 16 + ln;
#pragma unroll
        for (int i = 0; i < 8; ++i) bfr.u[i] = smKV[n * 32 + ks * 16 + i + hf * 8];
        acc[nf].v = __builtin_amdgcn_wmma_f32_16x16x32_bf16(
            false, afr.v, false, bfr.v, (short)0, acc[nf].v, false, false);
      }
    }
  }

  float Mr[8], ssum[8];
#pragma unroll
  for (int r = 0; r < 8; ++r) {
    Mr[r] = lfc_row[16 * w + r + 8 * hf] + m0v;   // sdc
    ssum[r] = 0.f;
  }

  const int ntiles = row0 / 64 + 1;
  for (int jt = 0; jt < ntiles; ++jt) {
    const int col0 = jt * 64;
    __syncthreads();
    for (int i = tid; i < 64; i += 128) {
      lfc_col[i] = lfcg[baseRow + col0 + i];
      ep_col[i] = epg[(size_t)b * S_ + col0 + i];
    }

    // ---- scores over full dh, streamed in 128-chunks (async staging) ----
    FragF sc[4];
#pragma unroll
    for (int nf = 0; nf < 4; ++nf)
#pragma unroll
      for (int r = 0; r < 8; ++r) sc[nf].f[r] = 0.f;

    for (int dc = 0; dc < DH_; dc += 128) {
      __syncthreads();
      for (int c = tid; c < 1024; c += 128) {
        int i = c >> 4, seg = c & 15;
        async_copy16((char*)smQ + i * 256 + seg * 16,
                     qb + (size_t)(row0 + i) * DH_ + dc + seg * 8);
      }
      for (int c = tid; c < 1024; c += 128) {
        int j = c >> 4, seg = c & 15;
        async_copy16((char*)smKV + j * 256 + seg * 16,
                     kb + (size_t)(col0 + j) * DH_ + dc + seg * 8);
      }
      async_wait0();
      __syncthreads();
#pragma unroll
      for (int ks = 0; ks < 4; ++ks) {
        Frag16 afr;
        int m = w * 16 + ln;
#pragma unroll
        for (int i = 0; i < 8; ++i) afr.u[i] = smQ[m * 64 + ks * 16 + i + hf * 8];
#pragma unroll
        for (int nf = 0; nf < 4; ++nf) {
          Frag16 bfr;
          int j = nf * 16 + ln;
#pragma unroll
          for (int i = 0; i < 8; ++i) bfr.u[i] = smKV[j * 64 + ks * 16 + i + hf * 8];
          sc[nf].v = __builtin_amdgcn_wmma_f32_16x16x32_bf16(
              false, afr.v, false, bfr.v, (short)0, sc[nf].v, false, false);
        }
      }
    }

    // ---- masking, online max, rescale, P tile ----
    float rmax[8];
#pragma unroll
    for (int r = 0; r < 8; ++r) rmax[r] = -__builtin_inff();
#pragma unroll
    for (int f = 0; f < 4; ++f) {
      int jl = f * 16 + ln, jg = col0 + jl;
#pragma unroll
      for (int r = 0; r < 8; ++r) {
        int il = 16 * w + r + 8 * hf, ig_ = row0 + il;
        bool ok = (jg <= ig_) && (ep_col[jl] == ep_row[il]);
        float lg = ok ? (lfc_row[il] - lfc_col[jl] + gi_row[il]) : -__builtin_inff();
        rmax[r] = fmaxf(rmax[r], lg);
      }
    }
#pragma unroll
    for (int r = 0; r < 8; ++r)
#pragma unroll
      for (int mm = 1; mm < 16; mm <<= 1)
        rmax[r] = fmaxf(rmax[r], __shfl_xor(rmax[r], mm, 16));

    float Mn[8], scale[8];
#pragma unroll
    for (int r = 0; r < 8; ++r) {
      Mn[r] = fmaxf(Mr[r], rmax[r]);
      scale[r] = expf(Mr[r] - Mn[r]);
      Mr[r] = Mn[r];
    }
#pragma unroll
    for (int nf = 0; nf < 16; ++nf)
#pragma unroll
      for (int r = 0; r < 8; ++r) acc[nf].f[r] *= scale[r];

    float rp[8] = {0, 0, 0, 0, 0, 0, 0, 0};
#pragma unroll
    for (int f = 0; f < 4; ++f) {
      int jl = f * 16 + ln, jg = col0 + jl;
#pragma unroll
      for (int r = 0; r < 8; ++r) {
        int il = 16 * w + r + 8 * hf, ig_ = row0 + il;
        bool ok = (jg <= ig_) && (ep_col[jl] == ep_row[il]);
        float p = 0.f;
        if (ok) {
          float lg = lfc_row[il] - lfc_col[jl] + gi_row[il];
          p = sc[f].f[r] * expf(lg - Mn[r]);
        }
        smP[w * 1024 + (r + 8 * hf) * 64 + jl] = f2bf(p);
        rp[r] += p;
      }
    }
#pragma unroll
    for (int r = 0; r < 8; ++r) {
#pragma unroll
      for (int mm = 1; mm < 16; mm <<= 1) rp[r] += __shfl_xor(rp[r], mm, 16);
      ssum[r] = ssum[r] * scale[r] + rp[r];
    }

    // ---- PV: acc += P @ V (dh half streamed in two 128-chunks) ----
    const unsigned* Pu = (const unsigned*)(smP + w * 1024);
    for (int dc = 0; dc < 2; ++dc) {
      __syncthreads();
      for (int p = tid; p < 128 * 32; p += 128) {
        int d = p >> 5, jp = p & 31;
        const unsigned short* vp = vb + (size_t)(col0 + jp * 2) * DH_ + dhalf + dc * 128 + d;
        smKV[4096 + d * 32 + jp] = (unsigned)vp[0] | ((unsigned)vp[DH_] << 16);
      }
      __syncthreads();
#pragma unroll
      for (int ks = 0; ks < 2; ++ks) {
        Frag16 afr;
#pragma unroll
        for (int i = 0; i < 8; ++i) afr.u[i] = Pu[ln * 32 + ks * 16 + i + hf * 8];
#pragma unroll
        for (int nf = 0; nf < 8; ++nf) {
          Frag16 bfr;
          int d = nf * 16 + ln;
#pragma unroll
          for (int i = 0; i < 8; ++i)
            bfr.u[i] = smKV[4096 + d * 32 + ks * 16 + i + hf * 8];
          acc[dc * 8 + nf].v = __builtin_amdgcn_wmma_f32_16x16x32_bf16(
              false, afr.v, false, bfr.v, (short)0, acc[dc * 8 + nf].v, false, false);
        }
      }
    }
  }

  // ---- epilogue: exact norm and output ----
#pragma unroll
  for (int r = 0; r < 8; ++r) {
    int il = 16 * w + r + 8 * hf;
    float stab = Mr[r];
    float sdc = lfc_row[il] + m0v;
    float internv = expf(sdc - stab) * qn_row[il];
    float normv = fmaxf(fabsf(ssum[r]) + internv, expf(-stab));
    float inv = 1.f / (normv + 1e-6f);
    size_t outbase = (baseRow + (size_t)(row0 + il)) * DH_ + dhalf;
#pragma unroll
    for (int nf = 0; nf < 16; ++nf)
      ht[outbase + nf * 16 + ln] = acc[nf].f[r] * inv;
  }
}

// ---------------------------------------------------------------------------
// h = (h_tilde + skip*conv_act) * silu(z)  -> bf16 for the down GEMM
// ---------------------------------------------------------------------------
__global__ __launch_bounds__(256) void epilogue_kernel(
    const float* __restrict__ ht, const float* __restrict__ conv_act,
    const float* __restrict__ up, const float* __restrict__ skip,
    unsigned short* __restrict__ hbf) {
  size_t idx = (size_t)blockIdx.x * 256 + threadIdx.x;
  int c = (int)(idx % HID_);
  size_t bt = idx / HID_;
  float hv = ht[idx] + skip[c] * conv_act[idx];
  float z = up[bt * (2 * HID_) + HID_ + c];
  hbf[idx] = f2bf(hv * siluf(z));
}

// ---------------------------------------------------------------------------
extern "C" void kernel_launch(void* const* d_in, const int* in_sizes, int n_in,
                              void* d_out, int out_size, void* d_ws, size_t ws_size,
                              hipStream_t stream) {
  (void)in_sizes; (void)n_in; (void)out_size; (void)ws_size;
  const float* inputs     = (const float*)d_in[0];
  const int*   mask       = (const int*)d_in[1];
  const float* conv_state = (const float*)d_in[2];
  const float* C0         = (const float*)d_in[3];
  const float* n0         = (const float*)d_in[4];
  const float* m0         = (const float*)d_in[5];
  const float* W_up       = (const float*)d_in[6];
  const float* conv_w     = (const float*)d_in[7];
  const float* conv_b     = (const float*)d_in[8];
  const float* Wq         = (const float*)d_in[9];
  const float* Wk         = (const float*)d_in[10];
  const float* Wv         = (const float*)d_in[11];
  const float* wi_k       = (const float*)d_in[12];
  const float* wi_b       = (const float*)d_in[13];
  const float* wf_k       = (const float*)d_in[14];
  const float* wf_b       = (const float*)d_in[15];
  const float* skip       = (const float*)d_in[16];
  const float* W_down     = (const float*)d_in[17];

  char* ws = (char*)d_ws;
  size_t off = 0;
  auto carve = [&](size_t bytes) -> void* {
    void* p = ws + off;
    off += (bytes + 255) & ~(size_t)255;
    return p;
  };
  float* up        = (float*)carve((size_t)B_ * S_ * 2 * HID_ * 4);
  float* conv_act  = (float*)carve((size_t)B_ * S_ * HID_ * 4);
  unsigned short* qbf = (unsigned short*)carve((size_t)B_ * S_ * HID_ * 2);
  unsigned short* kbf = (unsigned short*)carve((size_t)B_ * S_ * HID_ * 2);
  unsigned short* vbf = (unsigned short*)carve((size_t)B_ * S_ * HID_ * 2);
  float* ig   = (float*)carve((size_t)B_ * S_ * NH_ * 4);
  float* fg   = (float*)carve((size_t)B_ * S_ * NH_ * 4);
  float* lfc  = (float*)carve((size_t)B_ * NH_ * S_ * 4);
  int*   ep   = (int*)carve((size_t)B_ * S_ * 4);
  float* qn   = (float*)carve((size_t)B_ * NH_ * S_ * 4);
  float* ht   = (float*)carve((size_t)B_ * S_ * HID_ * 4);
  unsigned short* hbf    = (unsigned short*)carve((size_t)B_ * S_ * HID_ * 2);
  unsigned short* WupT   = (unsigned short*)carve((size_t)FEAT_ * 2 * HID_ * 2);
  unsigned short* WdownT = (unsigned short*)carve((size_t)HID_ * FEAT_ * 2);

  // 0) weight pre-transpose + bf16 convert (once per launch)
  transpose_bf16_kernel<<<(2 * HID_ * FEAT_) / 256, 256, 0, stream>>>(
      W_up, WupT, FEAT_, 2 * HID_);
  transpose_bf16_kernel<<<(FEAT_ * HID_) / 256, 256, 0, stream>>>(
      W_down, WdownT, HID_, FEAT_);
  // 1) up = inputs @ W_up  (4096 x 4096 x 1024)
  wmma_gemm_f32a<<<dim3((2 * HID_) / 128, (B_ * S_) / 128), 256, 0, stream>>>(
      inputs, WupT, up, B_ * S_, 2 * HID_, FEAT_);
  // 2) conv + silu + blockdiag qkv + gate pre-activations
  conv_qkv_gates<<<B_ * S_, 256, 0, stream>>>(
      up, conv_state, conv_w, conv_b, Wq, Wk, Wv,
      wi_k, wi_b, wf_k, wf_b, conv_act, qbf, kbf, vbf, ig, fg);
  // 3) scans
  scan_kernel<<<B_ * NH_, 32, 0, stream>>>(fg, mask, lfc, ep);
  // 4) qn = qh . n0
  qn_kernel<<<(B_ * NH_ * S_) / 8, 256, 0, stream>>>(qbf, n0, qn);
  // 5) fused flash-style mLSTM attention
  attn_kernel<<<dim3(S_ / 64, B_ * NH_, 2), 128, 0, stream>>>(
      qbf, kbf, vbf, lfc, ig, ep, qn, C0, m0, ht);
  // 6) elementwise epilogue -> bf16 h
  epilogue_kernel<<<(B_ * S_ * HID_) / 256, 256, 0, stream>>>(
      ht, conv_act, up, skip, hbf);
  // 7) y = h @ W_down  (4096 x 1024 x 2048) -> d_out
  wmma_gemm_bf16<<<dim3(FEAT_ / 128, (B_ * S_) / 128), 256, 0, stream>>>(
      hbf, WdownT, (float*)d_out, B_ * S_, FEAT_, HID_);
}
